// ContextAttention_56169582297516
// MI455X (gfx1250) — compile-verified, matrix operand order checked by reading, not run on hardware
//
#include <hip/hip_runtime.h>
#include <hip/hip_bf16.h>

typedef __attribute__((ext_vector_type(16))) _Float16 v16h;
typedef __attribute__((ext_vector_type(8)))  _Float16 v8h;
typedef __attribute__((ext_vector_type(8)))  float    v8f;

#define LOG2E 1.4426950408889634f

__device__ __forceinline__ float fast_exp(float x) {
    return __builtin_amdgcn_exp2f(x * LOG2E);
}
// tanh(x) = 1 - 2/(exp2(2x*log2e)+1); saturates to +/-1 for large |x|.
__device__ __forceinline__ float fast_tanh(float x) {
    float e = __builtin_amdgcn_exp2f(x * (2.0f * LOG2E));
    return 1.0f - 2.0f * __builtin_amdgcn_rcpf(e + 1.0f);
}

// ---------------------------------------------------------------------------
// fp32 -> f16 conversion (elementwise)
__global__ void cvt_f16_kernel(const float* __restrict__ in,
                               _Float16* __restrict__ out, int n) {
    int i = blockIdx.x * 256 + threadIdx.x;
    if (i < n) out[i] = (_Float16)in[i];
}

// fp32 [rows][cols] -> f16 [cols][rows] (transpose + convert)
__global__ void cvt_f16_tr_kernel(const float* __restrict__ in,
                                  _Float16* __restrict__ out,
                                  int rows, int cols) {
    int i = blockIdx.x * 256 + threadIdx.x;
    if (i < rows * cols) {
        int r = i / cols, c = i - r * cols;
        out[c * rows + r] = (_Float16)in[i];
    }
}

// ---------------------------------------------------------------------------
// C[M x N] = A[M x K] @ Bt[N x K]^T (+ bias[n]), f16 inputs, f32 accumulate.
// One wave per 16x16 output tile; K stepped by 32 using v_wmma_f32_16x16x32_f16.
// A-matrix VGPR layout (16x32 f16): row = lane&15; half-wave selects K-subgroup.
// half index h in [0,16): K = kk + (h>>3)*16 + (lane>>4)*8 + (h&7)
// -> two contiguous 8-half (16B) segments per operand per step.
__global__ void wmma_gemm_abt_kernel(const _Float16* __restrict__ A, int lda,
                                     const _Float16* __restrict__ Bt, int ldb,
                                     const float* __restrict__ bias,
                                     float* __restrict__ C, int ldc, int K) {
    const int lane  = threadIdx.x;          // blockDim.x == 32 (one wave)
    const int lr    = lane & 15;
    const int half  = lane >> 4;            // 0 or 1
    const int kbase = half * 8;
    const int mBase = blockIdx.x * 16;
    const int nBase = blockIdx.y * 16;

    const v8h* Arow = (const v8h*)(A  + (size_t)(mBase + lr) * lda);
    const v8h* Brow = (const v8h*)(Bt + (size_t)(nBase + lr) * ldb);

    v8f c = {};
    for (int kk = 0; kk < K; kk += 32) {
        v8h a0 = Arow[(kk + kbase) >> 3];
        v8h a1 = Arow[(kk + 16 + kbase) >> 3];
        v8h b0 = Brow[(kk + kbase) >> 3];
        v8h b1 = Brow[(kk + 16 + kbase) >> 3];
        v16h a, b;
#pragma unroll
        for (int i = 0; i < 8; ++i) {
            a[i] = a0[i]; a[8 + i] = a1[i];
            b[i] = b0[i]; b[8 + i] = b1[i];
        }
        c = __builtin_amdgcn_wmma_f32_16x16x32_f16(
                false, a, false, b, (short)0, c, false, false);
    }

    if (bias) {
        float bv = bias[nBase + lr];
#pragma unroll
        for (int r = 0; r < 8; ++r) c[r] += bv;
    }
    // C/D layout: VGPR r holds M = r + 8*half, N = lane&15
    const int col = nBase + lr;
#pragma unroll
    for (int r = 0; r < 8; ++r) {
        int row = mBase + r + half * 8;
        C[(size_t)row * ldc + col] = c[r];
    }
}

// ---------------------------------------------------------------------------
// Fused Bahdanau scores: s[n,m] = w_b + sum_d w[d]*tanh(q[n,d]+k[m,d])
// 16x16 (n,m) tile per 256-thread block; q/k rows staged in LDS (stride 260
// floats -> bank offset 4 per row, no 64-bank conflicts on the k reads).
__global__ void scores_kernel(const float* __restrict__ q,
                              const float* __restrict__ k,
                              const float* __restrict__ w_w,
                              const float* __restrict__ w_b,
                              float* __restrict__ s) {
    __shared__ float qs[16][260];
    __shared__ float ks[16][260];
    __shared__ float ws[256];
    const int tid   = threadIdx.x;               // 256 threads
    const int nBase = blockIdx.x * 16;
    const int mBase = blockIdx.y * 16;

    for (int i = tid; i < 16 * 256; i += 256) {
        int r = i >> 8, c = i & 255;
        qs[r][c] = q[(size_t)(nBase + r) * 256 + c];
        ks[r][c] = k[(size_t)(mBase + r) * 256 + c];
    }
    ws[tid] = w_w[tid];
    __syncthreads();

    const int tn = tid >> 4, tm = tid & 15;
    float acc = w_b[0];
#pragma unroll 8
    for (int d = 0; d < 256; ++d)
        acc += ws[d] * fast_tanh(qs[tn][d] + ks[tm][d]);
    s[(size_t)(nBase + tn) * 1024 + (mBase + tm)] = acc;
}

// ---------------------------------------------------------------------------
// Per-row softmax stats over M=1024: rmax[row], rinv[row] = 1/sum(exp)
__global__ void rowstats_kernel(const float* __restrict__ s,
                                float* __restrict__ rmax,
                                float* __restrict__ rinv) {
    __shared__ float red[256];
    const int row = blockIdx.x, tid = threadIdx.x;
    const float* sr = s + (size_t)row * 1024;

    float m = -1e30f;
    for (int j = tid; j < 1024; j += 256) m = fmaxf(m, sr[j]);
    red[tid] = m; __syncthreads();
    for (int st = 128; st > 0; st >>= 1) {
        if (tid < st) red[tid] = fmaxf(red[tid], red[tid + st]);
        __syncthreads();
    }
    m = red[0]; __syncthreads();

    float sum = 0.0f;
    for (int j = tid; j < 1024; j += 256) sum += fast_exp(sr[j] - m);
    red[tid] = sum; __syncthreads();
    for (int st = 128; st > 0; st >>= 1) {
        if (tid < st) red[tid] += red[tid + st];
        __syncthreads();
    }
    if (tid == 0) { rmax[row] = m; rinv[row] = __builtin_amdgcn_rcpf(red[0]); }
}

// P16[n,m] = f16( exp(s - rmax[n]) * rinv[n] )
__global__ void make_p16_kernel(const float* __restrict__ s,
                                const float* __restrict__ rmax,
                                const float* __restrict__ rinv,
                                _Float16* __restrict__ P) {
    int i = blockIdx.x * 256 + threadIdx.x;
    int row = i >> 10;
    float e = fast_exp(s[i] - rmax[row]) * rinv[row];
    P[i] = (_Float16)e;
}

// ---------------------------------------------------------------------------
// Stage-2 pooling: t[n] = ctx[n,:].wp_w + wp_b; alpha_p = softmax_n(t);
// out[d] = sum_n alpha_p[n]*ctx[n,d].  Single block, 1024 threads.
__global__ void final_pool_kernel(const float* __restrict__ ctx,
                                  const float* __restrict__ wp_w,
                                  const float* __restrict__ wp_b,
                                  float* __restrict__ out) {
    __shared__ float t[1024];
    __shared__ float red[1024];
    const int tid = threadIdx.x;

    float acc = wp_b[0];
    const float* cr = ctx + (size_t)tid * 256;
#pragma unroll 8
    for (int d = 0; d < 256; ++d) acc += cr[d] * wp_w[d];
    t[tid] = acc; red[tid] = acc; __syncthreads();

    for (int st = 512; st > 0; st >>= 1) {
        if (tid < st) red[tid] = fmaxf(red[tid], red[tid + st]);
        __syncthreads();
    }
    const float m = red[0]; __syncthreads();

    float e = fast_exp(t[tid] - m);
    t[tid] = e; red[tid] = e; __syncthreads();
    for (int st = 512; st > 0; st >>= 1) {
        if (tid < st) red[tid] += red[tid + st];
        __syncthreads();
    }
    const float inv = __builtin_amdgcn_rcpf(red[0]);
    __syncthreads();

    if (tid < 256) {
        float p = 0.0f;
        for (int n = 0; n < 1024; ++n) p += t[n] * ctx[(size_t)n * 256 + tid];
        out[tid] = p * inv;
    }
}

// ---------------------------------------------------------------------------
extern "C" void kernel_launch(void* const* d_in, const int* in_sizes, int n_in,
                              void* d_out, int out_size, void* d_ws, size_t ws_size,
                              hipStream_t stream) {
    const int N = 1024, M = 1024, D = 256;

    const float* f_r   = (const float*)d_in[0];
    const float* f_rp  = (const float*)d_in[1];
    const float* W_w   = (const float*)d_in[2];
    const float* W_b   = (const float*)d_in[3];
    const float* Wp_w  = (const float*)d_in[4];
    const float* Wp_b  = (const float*)d_in[5];
    const float* w_w   = (const float*)d_in[6];
    const float* w_b   = (const float*)d_in[7];
    const float* wp_w  = (const float*)d_in[8];
    const float* wp_b  = (const float*)d_in[9];
    float* out = (float*)d_out;

    // workspace carve-up (256B aligned pieces)
    char* p = (char*)d_ws;
    auto take = [&](size_t bytes) { char* r = p; p += (bytes + 255) & ~size_t(255); return r; };
    float*    q      = (float*)   take((size_t)N * D * 4);   // [N,D]
    float*    kmat   = (float*)   take((size_t)M * D * 4);   // [M,D]
    _Float16* fr16   = (_Float16*)take((size_t)N * D * 2);   // [N,D]
    _Float16* frp16  = (_Float16*)take((size_t)M * D * 2);   // [M,D]
    _Float16* W16    = (_Float16*)take((size_t)D * D * 2);   // [D,D]
    _Float16* Wp16   = (_Float16*)take((size_t)D * D * 2);   // [D,D]
    _Float16* FT16   = (_Float16*)take((size_t)D * M * 2);   // [D,M] = f_rp^T
    float*    s      = (float*)   take((size_t)N * M * 4);   // [N,M]
    float*    rmax   = (float*)   take((size_t)N * 4);
    float*    rinv   = (float*)   take((size_t)N * 4);
    _Float16* P16    = (_Float16*)take((size_t)N * M * 2);   // [N,M]
    float*    ctx    = (float*)   take((size_t)N * D * 4);   // [N,D]

    // 1) precision conversion / transpose
    cvt_f16_kernel<<<dim3((N * D + 255) / 256), 256, 0, stream>>>(f_r,  fr16,  N * D);
    cvt_f16_kernel<<<dim3((M * D + 255) / 256), 256, 0, stream>>>(f_rp, frp16, M * D);
    cvt_f16_kernel<<<dim3((D * D + 255) / 256), 256, 0, stream>>>(W_w,  W16,   D * D);
    cvt_f16_kernel<<<dim3((D * D + 255) / 256), 256, 0, stream>>>(Wp_w, Wp16,  D * D);
    cvt_f16_tr_kernel<<<dim3((M * D + 255) / 256), 256, 0, stream>>>(f_rp, FT16, M, D);

    // 2) projections via WMMA: q = fr16 @ W16^T + W_b ; k = frp16 @ Wp16^T + Wp_b
    wmma_gemm_abt_kernel<<<dim3(N / 16, D / 16), 32, 0, stream>>>(
        fr16, D, W16, D, W_b, q, D, D);
    wmma_gemm_abt_kernel<<<dim3(M / 16, D / 16), 32, 0, stream>>>(
        frp16, D, Wp16, D, Wp_b, kmat, D, D);

    // 3) fused tanh-score tensor (the transcendental-bound core)
    scores_kernel<<<dim3(N / 16, M / 16), 256, 0, stream>>>(q, kmat, w_w, w_b, s);

    // 4) row softmax stats + f16 probability matrix
    rowstats_kernel<<<dim3(N), 256, 0, stream>>>(s, rmax, rinv);
    make_p16_kernel<<<dim3((N * M) / 256), 256, 0, stream>>>(s, rmax, rinv, P16);

    // 5) context = P16 @ f_rp via WMMA (Bt = f_rp^T, contiguous in K)
    wmma_gemm_abt_kernel<<<dim3(N / 16, D / 16), 32, 0, stream>>>(
        P16, M, FT16, M, nullptr, ctx, D, M);

    // 6) stage-2 attention pooling -> out[0:256]
    final_pool_kernel<<<dim3(1), 1024, 0, stream>>>(ctx, wp_w, wp_b, out);
}
